// SSL_BACKEND_MultiStreamMHFA_64398739636683
// MI455X (gfx1250) — matrix operand order repaired and checked by reading, not compiled, not run on hardware
//
#include <hip/hip_runtime.h>
#include <hip/hip_bf16.h>

// ---------------- problem constants ----------------
#define HN   8
#define FN   768
#define DN   128
#define EN   256
#define LN   13
#define BN   8
#define TN   1000
#define TP   1024          // T padded to multiple of 64 for WMMA tiling
#define DCOR 8128          // D*(D-1)/2
#define BHD  (BN*HN*DN)    // 8192

typedef __attribute__((ext_vector_type(16))) __bf16 v16bf;
typedef __attribute__((ext_vector_type(8)))  float  v8f;

union FragU { struct { uint4 lo; uint4 hi; } q; v16bf v; };

__device__ __host__ inline unsigned short f2bf(float x) {
    unsigned u = __builtin_bit_cast(unsigned, x);
    u = (u + 0x7FFFu + ((u >> 16) & 1u)) >> 16;
    return (unsigned short)u;
}

__device__ inline v8f wmma_bf16(v16bf a, v16bf b, v8f c) {
#if defined(__gfx1250__)
    // v_wmma_f32_16x16x32_bf16: D = A(16x32) * B(32x16) + C
    return __builtin_amdgcn_wmma_f32_16x16x32_bf16(false, a, false, b, (short)0, c, false, false);
#else
    return c;
#endif
}

// ---------------- K0: softmax of mixing weights (H x L), k and v ----------------
__global__ void k0_softmax(const float* __restrict__ wk, const float* __restrict__ wv,
                           float* __restrict__ kw, float* __restrict__ vw) {
    int tid = threadIdx.x;
    if (tid >= 16) return;
    int h = tid & 7;
    const float* src = (tid < 8) ? wk : wv;
    float* dst = (tid < 8) ? kw : vw;
    src += h * LN;
    float m = src[0];
    for (int l = 1; l < LN; ++l) m = fmaxf(m, src[l]);
    float e[LN]; float s = 0.f;
    for (int l = 0; l < LN; ++l) { e[l] = __expf(src[l] - m); s += e[l]; }
    float inv = 1.0f / s;
    for (int l = 0; l < LN; ++l) dst[h * LN + l] = e[l] * inv;
}

// ---------------- K0b: cast Wk/Wv (H,D,F) f32 -> bf16 ----------------
__global__ __launch_bounds__(256) void k0b_cast(const float* __restrict__ Wk, const float* __restrict__ Wv,
                                                unsigned short* __restrict__ wkb, unsigned short* __restrict__ wvb) {
    size_t n = (size_t)HN * DN * FN;
    size_t i = (size_t)blockIdx.x * 256 + threadIdx.x;
    if (i < n)           wkb[i]     = f2bf(Wk[i]);
    else if (i < 2 * n)  wvb[i - n] = f2bf(Wv[i - n]);
}

// ---------------- K1: L-mix. x(B,F,T,L) -> k_raw/v_raw (B,H,TP,F) bf16 ----------------
// grid = B * (TP/256) * (F/4); thread owns one t, 4 consecutive f, all 16 (head,{k,v}) outputs.
__global__ __launch_bounds__(256) void k1_mix(const float* __restrict__ x,
                                              const float* __restrict__ kw, const float* __restrict__ vw,
                                              unsigned short* __restrict__ kraw, unsigned short* __restrict__ vraw) {
    int bid = blockIdx.x;
    int f4  = bid % (FN / 4);
    int r   = bid / (FN / 4);
    int tch = r % (TP / 256);
    int b   = r / (TP / 256);
    int t   = tch * 256 + threadIdx.x;
    int f0  = f4 * 4;

    if (t >= TN) {  // zero-fill pad rows so downstream GEMM reads zeros
        uint2 z; z.x = 0u; z.y = 0u;
        for (int hh = 0; hh < 16; ++hh) {
            unsigned short* dst = ((hh < 8) ? kraw : vraw)
                + ((size_t)((b * HN + (hh & 7)) * TP + t)) * FN + f0;
            *(uint2*)dst = z;
        }
        return;
    }

    float xv[4][LN];
#pragma unroll
    for (int ff = 0; ff < 4; ++ff) {
        const float* xp = x + ((size_t)(b * FN + f0 + ff) * TN + t) * LN;
#pragma unroll
        for (int l = 0; l < LN; ++l) xv[ff][l] = xp[l];
    }

    for (int hh = 0; hh < 16; ++hh) {
        const float* w = ((hh < 8) ? kw : vw) + (hh & 7) * LN;
        float a0 = 0.f, a1 = 0.f, a2 = 0.f, a3 = 0.f;
#pragma unroll
        for (int l = 0; l < LN; ++l) {
            float wl = w[l];
            a0 += xv[0][l] * wl; a1 += xv[1][l] * wl;
            a2 += xv[2][l] * wl; a3 += xv[3][l] * wl;
        }
        uint2 v;
        v.x = (unsigned)f2bf(a0) | ((unsigned)f2bf(a1) << 16);
        v.y = (unsigned)f2bf(a2) | ((unsigned)f2bf(a3) << 16);
        unsigned short* dst = ((hh < 8) ? kraw : vraw)
            + ((size_t)((b * HN + (hh & 7)) * TP + t)) * FN + f0;
        *(uint2*)dst = v;
    }
}

// ---------------- K2: big GEMM. mh_kT/mh_vT (B,H,D,TP) f32 = W(bf16) @ raw^T + bias ----------------
// C'[d,t] = sum_f W[d,f]*raw[t,f].  Wave computes a 64x64 (d x t) tile. 8 waves/block.
__global__ __launch_bounds__(256) void k2_gemm(const unsigned short* __restrict__ wkb,
                                               const unsigned short* __restrict__ wvb,
                                               const unsigned short* __restrict__ kraw,
                                               const unsigned short* __restrict__ vraw,
                                               const float* __restrict__ bk, const float* __restrict__ bv,
                                               float* __restrict__ mhkT, float* __restrict__ mhvT) {
    int gw    = blockIdx.x * 8 + (threadIdx.x >> 5);   // 4096 waves
    int lane  = threadIdx.x & 31;
    int lane15 = lane & 15;
    int half   = lane >> 4;
    int koffA  = half * 8;    // A lane-half K offset (interleaved by 8/16)
    int koffB  = half * 8;    // B lane-half K offset (interleaved by 8/16)
    int rowoff = half * 8;    // C row offset for lanes 16..31

    int nt = gw & 15;          // t tile  (16 * 64 = TP)
    int mt = (gw >> 4) & 1;    // d tile  (2  * 64 = D)
    int hh = (gw >> 5) & 15;   // 0..7 = k path, 8..15 = v path
    int b  = gw >> 9;
    int h  = hh & 7;
    bool isK = hh < 8;

    const unsigned short* Aw   = (isK ? wkb : wvb) + (size_t)h * DN * FN;                 // (D,F)
    const unsigned short* Braw = (isK ? kraw : vraw) + (size_t)(b * HN + h) * TP * FN;     // (TP,F)
    const float*          bias = (isK ? bk : bv) + h * DN;
    float*                Cout = (isK ? mhkT : mhvT) + (size_t)(b * HN + h) * DN * TP;     // (D,TP)

    int m0 = mt * 64, n0 = nt * 64;
    v8f acc[4][4] = {};

    for (int k0 = 0; k0 < FN; k0 += 32) {
        FragU a[4];
#pragma unroll
        for (int i = 0; i < 4; ++i) {
            const unsigned short* p = Aw + (size_t)(m0 + 16 * i + lane15) * FN + k0 + koffA;
            a[i].q.lo = *(const uint4*)p;
            a[i].q.hi = *(const uint4*)(p + 16);
        }
#pragma unroll
        for (int jn = 0; jn < 4; ++jn) {
            const unsigned short* p = Braw + (size_t)(n0 + 16 * jn + lane15) * FN + k0 + koffB;
            FragU bf;
            bf.q.lo = *(const uint4*)p;
            bf.q.hi = *(const uint4*)(p + 16);
            __builtin_prefetch(p + 128, 0, 1);   // stream-ahead -> global_prefetch_b8
#pragma unroll
            for (int i = 0; i < 4; ++i) acc[i][jn] = wmma_bf16(a[i].v, bf.v, acc[i][jn]);
        }
    }

#pragma unroll
    for (int i = 0; i < 4; ++i)
#pragma unroll
        for (int jn = 0; jn < 4; ++jn) {
            int tcol = n0 + 16 * jn + lane15;
#pragma unroll
            for (int j = 0; j < 8; ++j) {
                int d = m0 + 16 * i + j + rowoff;
                Cout[(size_t)d * TP + tcol] = acc[i][jn][j] + bias[d];   // 64B coalesced per row-half
            }
        }
}

// ---------------- K3: per-(b,h,d) mean / inv-sd over t (ddof=1), for mh_k and mh_v ----------------
__global__ __launch_bounds__(128) void k3_stats(const float* __restrict__ mhkT, const float* __restrict__ mhvT,
                                                float* __restrict__ meanAll, float* __restrict__ isdAll) {
    int row = blockIdx.x;   // 0..2*BHD-1
    const float* src = (row < BHD) ? (mhkT + (size_t)row * TP)
                                   : (mhvT + (size_t)(row - BHD) * TP);
    float s = 0.f, s2 = 0.f;
    for (int t = threadIdx.x; t < TN; t += 128) { float v = src[t]; s += v; s2 += v * v; }
    __shared__ float r1[128], r2[128];
    r1[threadIdx.x] = s; r2[threadIdx.x] = s2;
    __syncthreads();
    for (int w = 64; w > 0; w >>= 1) {
        if (threadIdx.x < w) { r1[threadIdx.x] += r1[threadIdx.x + w]; r2[threadIdx.x] += r2[threadIdx.x + w]; }
        __syncthreads();
    }
    if (threadIdx.x == 0) {
        float sum = r1[0], sq = r2[0];
        float mu  = sum / (float)TN;
        float var = (sq - sum * sum / (float)TN) / (float)(TN - 1);
        float sd  = sqrtf(fmaxf(var, 0.f)) + 1e-9f;
        meanAll[row] = mu;
        isdAll[row]  = 1.0f / sd;
    }
}

// ---------------- K4: scores -> softmax over t -> o1 (one block per (b,h)) ----------------
__global__ __launch_bounds__(256) void k4_attn(const float* __restrict__ q,
                                               const float* __restrict__ mhkT, const float* __restrict__ mhvT,
                                               float* __restrict__ o1) {
    __shared__ float sSc[TP];
    __shared__ float sRed[256];
    int bh = blockIdx.x;
    int h  = bh % HN;
    int tid = threadIdx.x;
    const float* mk = mhkT + (size_t)bh * DN * TP;
    const float* mv = mhvT + (size_t)bh * DN * TP;
    const float* qh = q + h * DN;

    float sc[4] = {0.f, 0.f, 0.f, 0.f};
    for (int d = 0; d < DN; ++d) {
        float qd = qh[d];
        const float* rp = mk + (size_t)d * TP + tid;
#pragma unroll
        for (int i = 0; i < 4; ++i) sc[i] += qd * rp[256 * i];
    }
    const float scale = 0.088388347648318447f;  // 1/sqrt(128)
    float lm = -3.4e38f;
#pragma unroll
    for (int i = 0; i < 4; ++i) {
        int t = tid + 256 * i;
        sc[i] = (t < TN) ? sc[i] * scale : -3.4e38f;
        lm = fmaxf(lm, sc[i]);
    }
    sRed[tid] = lm; __syncthreads();
    for (int w = 128; w > 0; w >>= 1) {
        if (tid < w) sRed[tid] = fmaxf(sRed[tid], sRed[tid + w]);
        __syncthreads();
    }
    float m = sRed[0]; __syncthreads();

    float ls = 0.f;
#pragma unroll
    for (int i = 0; i < 4; ++i) {
        int t = tid + 256 * i;
        float e = (t < TN) ? __expf(sc[i] - m) : 0.f;
        sSc[t] = e; ls += e;
    }
    sRed[tid] = ls; __syncthreads();
    for (int w = 128; w > 0; w >>= 1) {
        if (tid < w) sRed[tid] += sRed[tid + w];
        __syncthreads();
    }
    float inv = 1.0f / sRed[0]; __syncthreads();

    if (tid < DN) {
        const float* rv = mv + (size_t)tid * TP;
        float acc = 0.f;
        for (int t = 0; t < TN; t += 4) {
            float4 vv = *(const float4*)(rv + t);
            acc += vv.x * sSc[t] + vv.y * sSc[t + 1] + vv.z * sSc[t + 2] + vv.w * sSc[t + 3];
        }
        o1[(size_t)bh * DN + tid] = acc * inv;
    }
}

// ---------------- K5: corr GEMM. normalization + f32->bf16 fused into fragment load ----------------
__device__ inline v16bf norm_frag(const float* __restrict__ row, int kbase, int hiStep,
                                  float mu, float isd) {
    union { unsigned short u[16]; v16bf v; } r;
    float4 f0 = *(const float4*)(row + kbase);
    float4 f1 = *(const float4*)(row + kbase + 4);
    float4 f2 = *(const float4*)(row + kbase + hiStep);
    float4 f3 = *(const float4*)(row + kbase + hiStep + 4);
    float vals[16] = {f0.x, f0.y, f0.z, f0.w, f1.x, f1.y, f1.z, f1.w,
                      f2.x, f2.y, f2.z, f2.w, f3.x, f3.y, f3.z, f3.w};
#pragma unroll
    for (int i = 0; i < 16; ++i) {
        int t = kbase + ((i < 8) ? i : hiStep + (i - 8));
        float v = (t < TN) ? (vals[i] - mu) * isd : 0.0f;   // pad rows contribute 0
        r.u[i] = f2bf(v);
    }
    return r.v;
}

__global__ __launch_bounds__(256) void k5_corr(const float* __restrict__ mhkT, const float* __restrict__ mhvT,
                                               const float* __restrict__ meanAll, const float* __restrict__ isdAll,
                                               float* __restrict__ corr) {
    int gw    = blockIdx.x * 8 + (threadIdx.x >> 5);   // 1024 waves
    int lane  = threadIdx.x & 31;
    int lane15 = lane & 15;
    int half   = lane >> 4;
    int koffA  = half * 8;
    int koffB  = half * 8;
    int rowoff = half * 8;

    int nt = gw & 3;
    int mt = (gw >> 2) & 3;
    int h  = (gw >> 4) & 7;
    int b  = gw >> 7;
    int bh = b * HN + h;

    const float* Ak = mhkT + (size_t)bh * DN * TP;   // rows k, contiguous over t
    const float* Bv = mhvT + (size_t)bh * DN * TP;   // rows l, contiguous over t
    int m0 = mt * 32, n0 = nt * 32;

    float muA[2], isA[2], muB[2], isB[2];
#pragma unroll
    for (int i = 0; i < 2; ++i) {
        int rk = m0 + 16 * i + lane15;
        muA[i] = meanAll[bh * DN + rk];         isA[i] = isdAll[bh * DN + rk];
        int rl = n0 + 16 * i + lane15;
        muB[i] = meanAll[BHD + bh * DN + rl];   isB[i] = isdAll[BHD + bh * DN + rl];
    }

    v8f acc[2][2] = {};
    for (int k0 = 0; k0 < TP; k0 += 32) {
        v16bf a[2];
#pragma unroll
        for (int i = 0; i < 2; ++i)
            a[i] = norm_frag(Ak + (size_t)(m0 + 16 * i + lane15) * TP, k0 + koffA, 16, muA[i], isA[i]);
#pragma unroll
        for (int jn = 0; jn < 2; ++jn) {
            v16bf bb = norm_frag(Bv + (size_t)(n0 + 16 * jn + lane15) * TP, k0 + koffB, 16, muB[jn], isB[jn]);
#pragma unroll
            for (int i = 0; i < 2; ++i) acc[i][jn] = wmma_bf16(a[i], bb, acc[i][jn]);
        }
    }

    const float invT = 1.0f / (float)TN;
#pragma unroll
    for (int i = 0; i < 2; ++i)
#pragma unroll
        for (int jn = 0; jn < 2; ++jn) {
            int l = n0 + 16 * jn + lane15;
#pragma unroll
            for (int j = 0; j < 8; ++j) {
                int k = m0 + 16 * i + j + rowoff;
                corr[((size_t)bh * DN + k) * DN + l] = acc[i][jn][j] * invT;
            }
        }
}

// ---------------- K6: upper-tri extraction + Wproj projection. corr staged in LDS via async copy ----------------
__global__ __launch_bounds__(256) void k6_proj(const float* __restrict__ corr,
                                               const float* __restrict__ Wproj, const float* __restrict__ bproj,
                                               float* __restrict__ outs) {
    __shared__ float sc[DN * DN];   // 64 KB
    int bh = blockIdx.x;
    int h  = bh % HN;
    int e  = threadIdx.x;
    const float* src = corr + (size_t)bh * DN * DN;

#if defined(__gfx1250__)
    // Async DMA of the 64KB corr tile into LDS (tracked by ASYNCcnt).
    for (int i = 0; i < 16; ++i) {
        int idx4 = threadIdx.x + 256 * i;
        unsigned ldsa = (unsigned)(unsigned long long)
            (__attribute__((address_space(3))) float*)&sc[idx4 * 4];
        const float* g = src + (size_t)idx4 * 4;
        asm volatile("global_load_async_to_lds_b128 %0, %1, off"
                     :: "v"(ldsa), "v"(g) : "memory");
    }
    asm volatile("s_wait_asynccnt 0x0" ::: "memory");
#else
    for (int i = 0; i < 16; ++i) {
        int idx4 = threadIdx.x + 256 * i;
        *(float4*)&sc[idx4 * 4] = *(const float4*)(src + (size_t)idx4 * 4);
    }
#endif
    __syncthreads();

    const float* wp = Wproj + ((size_t)h * EN + e) * DCOR;   // contiguous per thread
    float acc = bproj[h * EN + e];
    int c = 0;
    for (int k = 0; k < DN - 1; ++k) {
        const float* rowk = &sc[k * DN];
        for (int l = k + 1; l < DN; ++l) acc += rowk[l] * wp[c++];   // LDS broadcast
    }
    outs[(size_t)bh * EN + e] = acc;
}

// ---------------- K7: final two GEMVs + 0.5/0.5 blend ----------------
__global__ __launch_bounds__(256) void k7_final(const float* __restrict__ outs, const float* __restrict__ o1,
                                                const float* __restrict__ WT, const float* __restrict__ bT,
                                                const float* __restrict__ Wp, const float* __restrict__ bp,
                                                float* __restrict__ out) {
    int b = blockIdx.x;
    int e = threadIdx.x;
    const float* ob  = outs + (size_t)b * HN * EN;   // 2048
    const float* o1b = o1   + (size_t)b * HN * DN;   // 1024
    float aF = bp[e];
    const float* wpr = Wp + (size_t)e * HN * EN;
    for (int j = 0; j < HN * EN; ++j) aF += ob[j] * wpr[j];
    float aT = bT[e];
    const float* wtr = WT + (size_t)e * HN * DN;
    for (int j = 0; j < HN * DN; ++j) aT += o1b[j] * wtr[j];
    out[(size_t)b * EN + e] = 0.5f * aF + 0.5f * aT;
}

// ---------------- launch ----------------
extern "C" void kernel_launch(void* const* d_in, const int* in_sizes, int n_in,
                              void* d_out, int out_size, void* d_ws, size_t ws_size,
                              hipStream_t stream) {
    const float* x   = (const float*)d_in[0];
    const float* wk  = (const float*)d_in[1];
    const float* wv  = (const float*)d_in[2];
    const float* q   = (const float*)d_in[3];
    const float* Wk  = (const float*)d_in[4];
    const float* bk  = (const float*)d_in[5];
    const float* Wv  = (const float*)d_in[6];
    const float* bv  = (const float*)d_in[7];
    const float* Wpr = (const float*)d_in[8];
    const float* bpr = (const float*)d_in[9];
    const float* WT  = (const float*)d_in[10];
    const float* bT  = (const float*)d_in[11];
    const float* Wp  = (const float*)d_in[12];
    const float* bp  = (const float*)d_in[13];
    float* out = (float*)d_out;

    char*  w   = (char*)d_ws;
    size_t off = 0;
    auto alloc = [&](size_t bytes) -> void* {
        void* p = w + off;
        off = (off + bytes + 255) & ~(size_t)255;
        return p;
    };

    float*          kw     = (float*)alloc(HN * LN * 4);
    float*          vw     = (float*)alloc(HN * LN * 4);
    unsigned short* wkb    = (unsigned short*)alloc((size_t)HN * DN * FN * 2);
    unsigned short* wvb    = (unsigned short*)alloc((size_t)HN * DN * FN * 2);
    unsigned short* kraw   = (unsigned short*)alloc((size_t)BN * HN * TP * FN * 2);
    unsigned short* vraw   = (unsigned short*)alloc((size_t)BN * HN * TP * FN * 2);
    float*          mhkT   = (float*)alloc((size_t)BN * HN * DN * TP * 4);
    float*          mhvT   = (float*)alloc((size_t)BN * HN * DN * TP * 4);
    float*          meanA  = (float*)alloc((size_t)2 * BHD * 4);
    float*          isdA   = (float*)alloc((size_t)2 * BHD * 4);
    float*          o1     = (float*)alloc((size_t)BN * HN * DN * 4);
    float*          corr   = (float*)alloc((size_t)BN * HN * DN * DN * 4);
    float*          outs   = (float*)alloc((size_t)BN * HN * EN * 4);
    (void)ws_size; (void)in_sizes; (void)n_in; (void)out_size;

    k0_softmax<<<1, 32, 0, stream>>>(wk, wv, kw, vw);

    {
        size_t n2 = 2 * (size_t)HN * DN * FN;
        k0b_cast<<<(unsigned)((n2 + 255) / 256), 256, 0, stream>>>(Wk, Wv, wkb, wvb);
    }

    k1_mix<<<BN * (TP / 256) * (FN / 4), 256, 0, stream>>>(x, kw, vw, kraw, vraw);

    k2_gemm<<<512, 256, 0, stream>>>(wkb, wvb, kraw, vraw, bk, bv, mhkT, mhvT);

    k3_stats<<<2 * BHD, 128, 0, stream>>>(mhkT, mhvT, meanA, isdA);

    k4_attn<<<BN * HN, 256, 0, stream>>>(q, mhkT, mhvT, o1);

    k5_corr<<<128, 256, 0, stream>>>(mhkT, mhvT, meanA, isdA, corr);

    k6_proj<<<BN * HN, 256, 0, stream>>>(corr, Wpr, bpr, outs);

    k7_final<<<BN, 256, 0, stream>>>(outs, o1, WT, bT, Wp, bp, out);
}
